// TransformerBlock_2121713844450
// MI455X (gfx1250) — matrix operand order repaired and checked by reading, not compiled, not run on hardware
//
#include <hip/hip_runtime.h>
#include <hip/hip_bf16.h>
#include <math.h>

// ---------------------------------------------------------------------------
// Transformer block (pre-LN, single-head causal attn w/ GLOBAL softmax, MLP)
// B=4 T=4096 D=512 F=2048, fp32 in/out.  All matmuls via bf16x3 split WMMA
// (hi+lo bf16 decomposition, 3x v_wmma_f32_16x16x32_bf16 per K=32 step,
// fp32 accumulate) => ~1e-4 accuracy at ~3x the rate of the f32 WMMA path.
// ---------------------------------------------------------------------------

typedef __attribute__((ext_vector_type(16))) __bf16        v16bf;
typedef __attribute__((ext_vector_type(8)))  float          v8f;
typedef __attribute__((ext_vector_type(8)))  unsigned short u16x8;

#define B_  4
#define T_  4096
#define D_  512
#define F_  2048
#define NT_ (B_ * T_)

// ---- fp32 -> (hi,lo) bf16 split (round-to-nearest-even) -------------------
__device__ inline void split2(float x, unsigned short& h, unsigned short& l) {
  unsigned int u = __float_as_uint(x);
  unsigned int r = u + 0x7fffu + ((u >> 16) & 1u);
  h = (unsigned short)(r >> 16);
  float hf = __uint_as_float(((unsigned int)h) << 16);
  float res = x - hf;
  unsigned int u2 = __float_as_uint(res);
  unsigned int r2 = u2 + 0x7fffu + ((u2 >> 16) & 1u);
  l = (unsigned short)(r2 >> 16);
}

// ---- WMMA operand loaders --------------------------------------------------
// A operand (16x32, row-major src [*, ldK]):
//   lane<16 : row m0+lane,    K = {kk..kk+7, kk+16..kk+23}
//   lane>=16: row m0+lane-16, K = {kk+8..kk+15, kk+24..kk+31}
// B operand (32x16, "Bt" row-major src [N, ldK], row n = output column):
//   lane<16 : col n0+lane,    K = {kk..kk+7, kk+16..kk+23}
//   lane>=16: col n0+lane-16, K = {kk+8..kk+15, kk+24..kk+31}
__device__ inline v16bf ldp(const unsigned short* p0, const unsigned short* p1) {
  union { u16x8 u[2]; v16bf v; } t;
  t.u[0] = *(const u16x8*)p0;
  t.u[1] = *(const u16x8*)p1;
  return t.v;
}
__device__ inline v16bf ld_a(const unsigned short* A, int ldK, int m0, int kk, int lane) {
  int r  = m0 + (lane & 15);
  int kb = kk + ((lane & 16) ? 8 : 0);
  const unsigned short* p = A + (size_t)r * ldK + kb;
  return ldp(p, p + 16);
}
__device__ inline v16bf ld_b(const unsigned short* Bt, int ldK, int n0, int kk, int lane) {
  int c  = n0 + (lane & 15);
  int kb = kk + ((lane & 16) ? 8 : 0);
  const unsigned short* p = Bt + (size_t)c * ldK + kb;
  return ldp(p, p + 16);
}
__device__ inline v8f wmma3(v16bf ah, v16bf al, v16bf bh, v16bf bl, v8f c) {
  c = __builtin_amdgcn_wmma_f32_16x16x32_bf16(false, al, false, bh, (short)0, c, false, false);
  c = __builtin_amdgcn_wmma_f32_16x16x32_bf16(false, ah, false, bl, (short)0, c, false, false);
  c = __builtin_amdgcn_wmma_f32_16x16x32_bf16(false, ah, false, bh, (short)0, c, false, false);
  return c;
}

// ---- weight split + transpose:  src fp32 [K,N]  ->  dst bf16 hi/lo [N,K] --
__global__ void splitw_kernel(const float* __restrict__ W,
                              unsigned short* __restrict__ oh,
                              unsigned short* __restrict__ ol, int K, int N) {
  int i = blockIdx.x * blockDim.x + threadIdx.x;
  if (i >= K * N) return;
  int k = i / N, n = i % N;
  unsigned short h, l; split2(W[i], h, l);
  oh[(size_t)n * K + k] = h;
  ol[(size_t)n * K + k] = l;
}

// ---- LayerNorm + split:  x [NT,512] -> hi/lo bf16 --------------------------
__global__ __launch_bounds__(256) void ln_split_kernel(
    const float* __restrict__ x, const float* __restrict__ g,
    const float* __restrict__ bt,
    unsigned short* __restrict__ oh, unsigned short* __restrict__ ol) {
  int row = blockIdx.x, t = threadIdx.x;
  size_t o = (size_t)row * D_;
  float v0 = x[o + t], v1 = x[o + t + 256];
  __shared__ float red[256];
  red[t] = v0 + v1; __syncthreads();
  for (int s = 128; s > 0; s >>= 1) { if (t < s) red[t] += red[t + s]; __syncthreads(); }
  float mu = red[0] * (1.0f / 512.0f); __syncthreads();
  float d0 = v0 - mu, d1 = v1 - mu;
  red[t] = d0 * d0 + d1 * d1; __syncthreads();
  for (int s = 128; s > 0; s >>= 1) { if (t < s) red[t] += red[t + s]; __syncthreads(); }
  float inv = rsqrtf(red[0] * (1.0f / 512.0f) + 1e-5f);
  float y0 = d0 * inv * g[t] + bt[t];
  float y1 = d1 * inv * g[t + 256] + bt[t + 256];
  unsigned short h0, l0, h1, l1; split2(y0, h0, l0); split2(y1, h1, l1);
  oh[o + t] = h0; ol[o + t] = l0; oh[o + t + 256] = h1; ol[o + t + 256] = l1;
}

// ---- Generic bf16x3 GEMM:  C[M,N] = A[M,K] * Bt[N,K]^T + bias --------------
// mode 1: split -> oh/ol [M,N]        (Q, K projections)
// mode 2: split -> oh/ol [N,M] (transposed store; V^T for attention)
// mode 3: exact GELU, split -> oh/ol [M,N]  (MLP hidden)
// mode 4: fp32 out = res + acc + bias  (final residual output)
// Block: 256 thr = 8 waves (4M x 2N); wave tile 16M x 64N (4 accum tiles).
__global__ __launch_bounds__(256) void gemm_bf3(
    const unsigned short* __restrict__ Ah, const unsigned short* __restrict__ Al,
    const unsigned short* __restrict__ Bh, const unsigned short* __restrict__ Bl,
    const float* __restrict__ bias, int M, int N, int K, int mode,
    unsigned short* __restrict__ oh, unsigned short* __restrict__ ol,
    float* __restrict__ of, const float* __restrict__ res) {
  int lane = threadIdx.x & 31;
  int w = threadIdx.x >> 5;
  int wm = w & 3, wn = w >> 2;
  int m0 = blockIdx.y * 64 + wm * 16;
  int n0 = blockIdx.x * 128 + wn * 64;
  v8f z = {0.f, 0.f, 0.f, 0.f, 0.f, 0.f, 0.f, 0.f};
  v8f acc[4] = {z, z, z, z};
  for (int kk = 0; kk < K; kk += 32) {
    v16bf ah = ld_a(Ah, K, m0, kk, lane);
    v16bf al = ld_a(Al, K, m0, kk, lane);
#pragma unroll
    for (int j = 0; j < 4; ++j) {
      v16bf bh = ld_b(Bh, K, n0 + j * 16, kk, lane);
      v16bf bl = ld_b(Bl, K, n0 + j * 16, kk, lane);
      acc[j] = wmma3(ah, al, bh, bl, acc[j]);
    }
  }
  int cn = lane & 15;
  int rbase = (lane & 16) ? 8 : 0;
#pragma unroll
  for (int j = 0; j < 4; ++j) {
    int col = n0 + j * 16 + cn;
    float bb = bias[col];
#pragma unroll
    for (int i = 0; i < 8; ++i) {
      int row = m0 + i + rbase;
      float v = acc[j][i] + bb;
      if (mode == 3) v = 0.5f * v * (1.0f + erff(v * 0.70710678118654752f));
      if (mode == 4) {
        of[(size_t)row * N + col] = res[(size_t)row * N + col] + v;
      } else if (mode == 2) {
        unsigned short hh, ll; split2(v, hh, ll);
        oh[(size_t)col * M + row] = hh;
        ol[(size_t)col * M + row] = ll;
      } else {
        unsigned short hh, ll; split2(v, hh, ll);
        oh[(size_t)row * N + col] = hh;
        ol[(size_t)row * N + col] = ll;
      }
    }
  }
}

// ---- Fused causal attention, single pass -----------------------------------
// Global softmax => exp(S)/sum(exp(S)), no max shift needed (|S*scale| < ~3).
// Block = 4 waves, one 16-row Q tile.  Per 128-token s-block:
//   phase 1: wave w computes S[16x32] for its s-slice (Q from LDS, K global),
//            exp+mask, accumulates Z partial, stores exp(S) hi/lo to LDS.
//   phase 2: wave w owns output cols [w*128, w*128+128): 8 fp32 accum tiles,
//            A = exp(S) from LDS, B = V^T from global.
__global__ __launch_bounds__(128) void attn_kernel(
    const unsigned short* __restrict__ qh, const unsigned short* __restrict__ ql,
    const unsigned short* __restrict__ kh, const unsigned short* __restrict__ kl,
    const unsigned short* __restrict__ vth, const unsigned short* __restrict__ vtl,
    float* __restrict__ attnO, float* __restrict__ zp) {
  __shared__ unsigned short sqh[16 * 512], sql[16 * 512];
  __shared__ unsigned short sph[16 * 128], spl[16 * 128];
  int b = blockIdx.y;
  int t0 = blockIdx.x * 16;
  int tid = threadIdx.x;
  int lane = tid & 31, w = tid >> 5;

  size_t qbase = ((size_t)b * T_ + t0) * D_;
  for (int i = tid; i < (16 * 512) / 8; i += 128) {  // 8 u16x8 chunks / thread
    int r = i >> 6, c = (i & 63) << 3;
    *(u16x8*)&sqh[r * 512 + c] = *(const u16x8*)&qh[qbase + (size_t)r * D_ + c];
    *(u16x8*)&sql[r * 512 + c] = *(const u16x8*)&ql[qbase + (size_t)r * D_ + c];
  }
  __syncthreads();

  const float scale = 0.04419417382415922f;  // 1/sqrt(512)
  v8f z = {0.f, 0.f, 0.f, 0.f, 0.f, 0.f, 0.f, 0.f};
  v8f oacc[8] = {z, z, z, z, z, z, z, z};
  int d0 = w * 128;
  int nsb = t0 / 128 + 1;
  float zpart = 0.f;

  for (int sb = 0; sb < nsb; ++sb) {
    int sbase = sb * 128;
    int sw = sbase + w * 32;
    // ---- phase 1: scores for s in [sw, sw+32) ----
    v8f s0 = z, s1 = z;
    for (int kk = 0; kk < D_; kk += 32) {
      v16bf ahq = ld_a(sqh, 512, 0, kk, lane);
      v16bf alq = ld_a(sql, 512, 0, kk, lane);
      v16bf bh0 = ld_b(kh, D_, b * T_ + sw, kk, lane);
      v16bf bl0 = ld_b(kl, D_, b * T_ + sw, kk, lane);
      s0 = wmma3(ahq, alq, bh0, bl0, s0);
      v16bf bh1 = ld_b(kh, D_, b * T_ + sw + 16, kk, lane);
      v16bf bl1 = ld_b(kl, D_, b * T_ + sw + 16, kk, lane);
      s1 = wmma3(ahq, alq, bh1, bl1, s1);
    }
    {
      int cn = lane & 15;
      int rbase = (lane & 16) ? 8 : 0;
#pragma unroll
      for (int i = 0; i < 8; ++i) {
        int m = i + rbase;
        int tg = t0 + m;
#pragma unroll
        for (int tile = 0; tile < 2; ++tile) {
          int sl = w * 32 + tile * 16 + cn;  // s within 128-block
          int sg = sbase + sl;
          float sv = tile ? s1[i] : s0[i];
          float pv = (sg <= tg) ? expf(sv * scale) : 0.f;
          zpart += pv;
          unsigned short hh, ll; split2(pv, hh, ll);
          sph[m * 128 + sl] = hh;
          spl[m * 128 + sl] = ll;
        }
      }
    }
    __syncthreads();
    // ---- phase 2: O += exp(S) @ V for this s-block ----
    for (int c = 0; c < 4; ++c) {
      v16bf pah = ld_a(sph, 128, 0, c * 32, lane);
      v16bf pal = ld_a(spl, 128, 0, c * 32, lane);
      int sg0 = b * T_ + sbase + c * 32;
#pragma unroll
      for (int j = 0; j < 8; ++j) {
        v16bf vbh = ld_b(vth, NT_, d0 + j * 16, sg0, lane);
        v16bf vbl = ld_b(vtl, NT_, d0 + j * 16, sg0, lane);
        oacc[j] = wmma3(pah, pal, vbh, vbl, oacc[j]);
      }
    }
    __syncthreads();
  }
  // deterministic Z partial per wave
  for (int off = 16; off > 0; off >>= 1) zpart += __shfl_xor(zpart, off, 32);
  if (lane == 0) zp[((size_t)b * (T_ / 16) + blockIdx.x) * 4 + w] = zpart;
  // write unnormalized O
  int cn = lane & 15;
  int rbase = (lane & 16) ? 8 : 0;
#pragma unroll
  for (int j = 0; j < 8; ++j)
#pragma unroll
    for (int i = 0; i < 8; ++i) {
      int row = t0 + i + rbase;
      attnO[((size_t)b * T_ + row) * D_ + d0 + j * 16 + cn] = oacc[j][i];
    }
}

// ---- Z = sum of per-wave partials (1024 per batch) -------------------------
__global__ __launch_bounds__(256) void reduce_z(const float* __restrict__ zp,
                                                float* __restrict__ Z) {
  __shared__ float red[256];
  int b = blockIdx.x, t = threadIdx.x;
  const float* p = zp + (size_t)b * 1024;
  red[t] = p[t] + p[t + 256] + p[t + 512] + p[t + 768];
  __syncthreads();
  for (int s = 128; s > 0; s >>= 1) { if (t < s) red[t] += red[t + s]; __syncthreads(); }
  if (t == 0) Z[b] = red[0];
}

// ---- x1 = x + attnO/Z[b];  h2 = LN2(x1) split -------------------------------
__global__ __launch_bounds__(256) void combine_ln2_kernel(
    const float* __restrict__ x, const float* __restrict__ ao,
    const float* __restrict__ Z, const float* __restrict__ g,
    const float* __restrict__ bt, float* __restrict__ x1,
    unsigned short* __restrict__ oh, unsigned short* __restrict__ ol) {
  int row = blockIdx.x, t = threadIdx.x;
  float invZ = 1.0f / Z[row >> 12];
  size_t o = (size_t)row * D_;
  float v0 = x[o + t] + ao[o + t] * invZ;
  float v1 = x[o + t + 256] + ao[o + t + 256] * invZ;
  x1[o + t] = v0; x1[o + t + 256] = v1;
  __shared__ float red[256];
  red[t] = v0 + v1; __syncthreads();
  for (int s = 128; s > 0; s >>= 1) { if (t < s) red[t] += red[t + s]; __syncthreads(); }
  float mu = red[0] * (1.0f / 512.0f); __syncthreads();
  float d0 = v0 - mu, d1 = v1 - mu;
  red[t] = d0 * d0 + d1 * d1; __syncthreads();
  for (int s = 128; s > 0; s >>= 1) { if (t < s) red[t] += red[t + s]; __syncthreads(); }
  float inv = rsqrtf(red[0] * (1.0f / 512.0f) + 1e-5f);
  float y0 = d0 * inv * g[t] + bt[t];
  float y1 = d1 * inv * g[t + 256] + bt[t + 256];
  unsigned short h0, l0, h1, l1; split2(y0, h0, l0); split2(y1, h1, l1);
  oh[o + t] = h0; ol[o + t] = l0; oh[o + t + 256] = h1; ol[o + t + 256] = l1;
}

// ===========================================================================
extern "C" void kernel_launch(void* const* d_in, const int* in_sizes, int n_in,
                              void* d_out, int out_size, void* d_ws, size_t ws_size,
                              hipStream_t stream) {
  const float* x    = (const float*)d_in[0];
  const float* Wq   = (const float*)d_in[1];
  const float* bq   = (const float*)d_in[2];
  const float* Wk   = (const float*)d_in[3];
  const float* bk   = (const float*)d_in[4];
  const float* Wv   = (const float*)d_in[5];
  const float* bv   = (const float*)d_in[6];
  const float* W1   = (const float*)d_in[7];
  const float* b1   = (const float*)d_in[8];
  const float* W2   = (const float*)d_in[9];
  const float* b2   = (const float*)d_in[10];
  const float* ln1g = (const float*)d_in[11];
  const float* ln1b = (const float*)d_in[12];
  const float* ln2g = (const float*)d_in[13];
  const float* ln2b = (const float*)d_in[14];

  char* p = (char*)d_ws;
  auto aus = [&](size_t n) { unsigned short* r = (unsigned short*)p;
                             p += ((n * 2 + 255) & ~(size_t)255); return r; };
  auto af  = [&](size_t n) { float* r = (float*)p;
                             p += ((n * 4 + 255) & ~(size_t)255); return r; };

  unsigned short* wqt_h = aus((size_t)D_ * D_);  unsigned short* wqt_l = aus((size_t)D_ * D_);
  unsigned short* wkt_h = aus((size_t)D_ * D_);  unsigned short* wkt_l = aus((size_t)D_ * D_);
  unsigned short* wvt_h = aus((size_t)D_ * D_);  unsigned short* wvt_l = aus((size_t)D_ * D_);
  unsigned short* w1t_h = aus((size_t)F_ * D_);  unsigned short* w1t_l = aus((size_t)F_ * D_);
  unsigned short* w2t_h = aus((size_t)D_ * F_);  unsigned short* w2t_l = aus((size_t)D_ * F_);
  unsigned short* h_h   = aus((size_t)NT_ * D_); unsigned short* h_l   = aus((size_t)NT_ * D_);
  unsigned short* q_h   = aus((size_t)NT_ * D_); unsigned short* q_l   = aus((size_t)NT_ * D_);
  unsigned short* k_h   = aus((size_t)NT_ * D_); unsigned short* k_l   = aus((size_t)NT_ * D_);
  unsigned short* vt_h  = aus((size_t)NT_ * D_); unsigned short* vt_l  = aus((size_t)NT_ * D_);
  unsigned short* h2_h  = aus((size_t)NT_ * D_); unsigned short* h2_l  = aus((size_t)NT_ * D_);
  unsigned short* g1_h  = aus((size_t)NT_ * F_); unsigned short* g1_l  = aus((size_t)NT_ * F_);
  float* attnO = af((size_t)NT_ * D_);
  float* x1    = af((size_t)NT_ * D_);
  float* zp    = af((size_t)B_ * (T_ / 16) * 4);
  float* Z     = af(4);

  // 1) split + transpose weights to bf16 hi/lo [N,K]
  splitw_kernel<<<(D_ * D_ + 255) / 256, 256, 0, stream>>>(Wq, wqt_h, wqt_l, D_, D_);
  splitw_kernel<<<(D_ * D_ + 255) / 256, 256, 0, stream>>>(Wk, wkt_h, wkt_l, D_, D_);
  splitw_kernel<<<(D_ * D_ + 255) / 256, 256, 0, stream>>>(Wv, wvt_h, wvt_l, D_, D_);
  splitw_kernel<<<(D_ * F_ + 255) / 256, 256, 0, stream>>>(W1, w1t_h, w1t_l, D_, F_);
  splitw_kernel<<<(F_ * D_ + 255) / 256, 256, 0, stream>>>(W2, w2t_h, w2t_l, F_, D_);

  // 2) LN1 -> h (hi/lo)
  ln_split_kernel<<<NT_, 256, 0, stream>>>(x, ln1g, ln1b, h_h, h_l);

  // 3) projections: q, k (row-major), v (transposed store)
  dim3 gp(D_ / 128, NT_ / 64);
  gemm_bf3<<<gp, 256, 0, stream>>>(h_h, h_l, wqt_h, wqt_l, bq, NT_, D_, D_, 1,
                                   q_h, q_l, nullptr, nullptr);
  gemm_bf3<<<gp, 256, 0, stream>>>(h_h, h_l, wkt_h, wkt_l, bk, NT_, D_, D_, 1,
                                   k_h, k_l, nullptr, nullptr);
  gemm_bf3<<<gp, 256, 0, stream>>>(h_h, h_l, wvt_h, wvt_l, bv, NT_, D_, D_, 2,
                                   vt_h, vt_l, nullptr, nullptr);

  // 4) fused attention (unnormalized O + Z partials), then Z reduce
  attn_kernel<<<dim3(T_ / 16, B_), 128, 0, stream>>>(q_h, q_l, k_h, k_l,
                                                     vt_h, vt_l, attnO, zp);
  reduce_z<<<B_, 256, 0, stream>>>(zp, Z);

  // 5) x1 = x + O/Z ; h2 = LN2(x1)
  combine_ln2_kernel<<<NT_, 256, 0, stream>>>(x, attnO, Z, ln2g, ln2b, x1, h2_h, h2_l);

  // 6) MLP: g1 = gelu(h2 @ W1 + b1) ; out = x1 + g1 @ W2 + b2
  gemm_bf3<<<dim3(F_ / 128, NT_ / 64), 256, 0, stream>>>(
      h2_h, h2_l, w1t_h, w1t_l, b1, NT_, F_, D_, 3, g1_h, g1_l, nullptr, nullptr);
  gemm_bf3<<<dim3(D_ / 128, NT_ / 64), 256, 0, stream>>>(
      g1_h, g1_l, w2t_h, w2t_l, b2, NT_, D_, F_, 4, nullptr, nullptr,
      (float*)d_out, x1);

  (void)in_sizes; (void)n_in; (void)out_size; (void)ws_size;
}
